// SimpleGNN_31293131719367
// MI455X (gfx1250) — matrix operand order repaired and checked by reading, not compiled, not run on hardware
//
#include <hip/hip_runtime.h>
#include <hip/hip_bf16.h>
#include <math.h>

// ---------------- problem constants ----------------
#define BG    128                 // graphs
#define NA    32                  // atoms per graph
#define NNODE (BG * NA)           // 4096
#define NEDGE (NNODE * NA)        // 131072
#define HD    256                 // hidden dim
#define TDIM  128                 // time dim
#define TYPED 100                 // type dim
#define NFREQ 10
#define EDGEP 96                  // edge feature dim padded 66 -> 96 (K mult of 32)
#define NLAY  4
#define EPSLN 1e-5f

typedef __attribute__((ext_vector_type(16))) _Float16 v16h;
typedef __attribute__((ext_vector_type(8)))  _Float16 v8h;
typedef __attribute__((ext_vector_type(8)))  float    v8f;

// fast silu: x * rcp(1+e^-x)  (v_rcp_f32; fine for the f16 pipeline)
__device__ __forceinline__ float siluf(float x) {
  return x * __builtin_amdgcn_rcpf(1.0f + __expf(-x));
}

__device__ __forceinline__ v8f wmma_f16(v16h a, v16h b, v8f c) {
  // D = A(16x32 f16) x B(32x16 f16) + C(16x16 f32)
  return __builtin_amdgcn_wmma_f32_16x16x32_f16(false, a, false, b, (short)0, c,
                                                false, false);
}

// A fragment (16x32 f16) from LDS tile with row stride 40 halves (16B aligned).
// Layout: lanes 0-15 row=lane hold K{0..15}; lanes 16-31 row=lane-16
// hold K{16..31} (matches swizzled B).
__device__ __forceinline__ v16h load_a_frag(const _Float16* As, int rowBase, int lane) {
  const int r  = rowBase + (lane & 15);
  const int kb = (lane < 16) ? 0 : 16;
  const v8h lo = *(const v8h*)(As + r * 40 + kb);
  const v8h hi = *(const v8h*)(As + r * 40 + kb + 8);
  v16h a;
#pragma unroll
  for (int h = 0; h < 8; ++h) { a[h] = lo[h]; a[8 + h] = hi[h]; }
  return a;
}

// B fragment from pre-swizzled weights in global memory. `bp` already includes
// this lane's +lane*16 offset, so loads are base+immediate (32B/lane, 1KB/wave).
__device__ __forceinline__ v16h load_b_frag_sw(const _Float16* __restrict__ bp) {
  const v8h lo = *(const v8h*)(bp);
  const v8h hi = *(const v8h*)(bp + 8);
  v16h b;
#pragma unroll
  for (int h = 0; h < 8; ++h) { b[h] = lo[h]; b[8 + h] = hi[h]; }
  return b;
}

// ---------------- weight cast / pad / swizzle ----------------
// dst layout: for K-chunk ck (32 rows) and N-subtile s (16 cols):
//   frag = dst + ((ck*(N/16) + s)*32)*16 ; element (k,n):
//   lane = (n%16) + 16*((k%32) >= 16), slot = k%16  ->  frag[lane*16 + slot]
// This is contiguous k order (k = 16*half + slot), same as load_a_frag, so the
// GEMM kernels can read fragments with two b128 loads per lane.
__global__ __launch_bounds__(256) void castswz_kernel(const float* __restrict__ src,
                                                      _Float16* __restrict__ dst,
                                                      int rows, int padK, int N) {
  int idx = blockIdx.x * blockDim.x + threadIdx.x;
  if (idx >= padK * N) return;
  int k = idx / N, n = idx - k * N;
  float v = (k < rows) ? src[k * N + n] : 0.0f;
  int ck = k >> 5, r = k & 31;
  int s = n >> 4;
  int lane = (n & 15) + ((r & 16) ? 16 : 0);
  int slot = r & 15;
  size_t di = (((size_t)(ck * (N >> 4) + s) * 32) + lane) * 16 + slot;
  dst[di] = (_Float16)v;
}

// ---------------- embeddings ----------------
__global__ __launch_bounds__(128) void time_emb_kernel(const float* __restrict__ t,
                                                       float* __restrict__ temb) {
  int b = blockIdx.x, k = threadIdx.x;                 // TDIM threads
  const float c = -logf(10000.0f) / 63.0f;             // half-1 = 63
  float v;
  if (k < 64) v = sinf(t[b] * __expf(c * (float)k));
  else        v = cosf(t[b] * __expf(c * (float)(k - 64)));
  temb[b * TDIM + k] = v;
}

__global__ __launch_bounds__(256) void type_emb_kernel(const float* __restrict__ at,
                                                       const float* __restrict__ Ws,
                                                       const float* __restrict__ bs,
                                                       float* __restrict__ te) {
  int node = blockIdx.x, h = threadIdx.x;
  float s = bs[h];
  const float* a = at + (size_t)node * TYPED;
  for (int k = 0; k < TYPED; ++k) s += a[k] * Ws[k * HD + h];
  te[(size_t)node * HD + h] = s;
}

__global__ __launch_bounds__(256) void node_emb_kernel(const float* __restrict__ te,
                                                       const float* __restrict__ temb,
                                                       const float* __restrict__ Wn,
                                                       const float* __restrict__ bn,
                                                       float* __restrict__ nf) {
  int node = blockIdx.x, h = threadIdx.x;
  int b = node >> 5;
  float s = bn[h];
  const float* tr = te + (size_t)node * HD;
  for (int k = 0; k < HD; ++k)   s += tr[k] * Wn[k * HD + h];
  const float* tm = temb + (size_t)b * TDIM;
  for (int k = 0; k < TDIM; ++k) s += tm[k] * Wn[(HD + k) * HD + h];
  nf[(size_t)node * HD + h] = s;
}

// ---------------- edge features (sin/cos frac embedding + l_polar), f16, padded ----
__global__ __launch_bounds__(256) void edge_feat_kernel(const float* __restrict__ frac,
                                                        const float* __restrict__ lpol,
                                                        _Float16* __restrict__ ef) {
  int e = blockIdx.x * blockDim.x + threadIdx.x;
  if (e >= NEDGE) return;
  int b = e >> 10, i = (e >> 5) & 31, j = e & 31;
  int ni = b * NA + i, nj = b * NA + j;
  _Float16* o = ef + (size_t)e * EDGEP;
  const float twopi = 6.283185307179586f;
#pragma unroll
  for (int k = 0; k < 3; ++k) {
    float d = frac[nj * 3 + k] - frac[ni * 3 + k];
    d -= floorf(d);                                    // mod(.,1)
    for (int f = 0; f < NFREQ; ++f) {
      float v = d * twopi * (float)f;
      o[k * NFREQ + f]       = (_Float16)sinf(v);
      o[30 + k * NFREQ + f]  = (_Float16)cosf(v);
    }
  }
#pragma unroll
  for (int p = 0; p < 6; ++p) o[60 + p] = (_Float16)lpol[b * 6 + p];
#pragma unroll
  for (int p = 66; p < EDGEP; ++p) o[p] = (_Float16)0.0f;
}

// ---------------- layernorm (row = 256) ----------------
__global__ __launch_bounds__(256) void ln_kernel(const float* __restrict__ x,
                                                 const float* __restrict__ g,
                                                 const float* __restrict__ b,
                                                 _Float16* __restrict__ o16,
                                                 float* __restrict__ o32) {
  __shared__ float red[256];
  int node = blockIdx.x, c = threadIdx.x;
  float v = x[(size_t)node * HD + c];
  red[c] = v; __syncthreads();
  for (int s = 128; s > 0; s >>= 1) { if (c < s) red[c] += red[c + s]; __syncthreads(); }
  float mu = red[0] * (1.0f / HD); __syncthreads();
  float d = v - mu;
  red[c] = d * d; __syncthreads();
  for (int s = 128; s > 0; s >>= 1) { if (c < s) red[c] += red[c + s]; __syncthreads(); }
  float var = red[0] * (1.0f / HD);
  float y = d * rsqrtf(var + EPSLN) * g[c] + b[c];
  if (o16) o16[(size_t)node * HD + c] = (_Float16)y;
  if (o32) o32[(size_t)node * HD + c] = y;
}

// ---------------- generic WMMA GEMM: out(M x 256) = silu?(A(MxK) @ Bw(Kx256) + bias)
// Block tile 128x128, 8 waves (4x2), wave tile 32x64, K chunk 32.
// Bw is pre-swizzled (fragment-major); B fragments load straight from global
// via a single advancing pointer + immediate offsets; all fragments preloaded
// before the WMMA burst so loadcnt waits retire progressively.
// Compile-time flags keep the epilogue branch-free.
template <bool HASBIAS, bool DOSILU, bool OUT16, bool OUT32>
__global__ __launch_bounds__(256) void gemm_wmma_kernel(
    const _Float16* __restrict__ A, int lda, int K,
    const _Float16* __restrict__ Bw, const float* __restrict__ bias,
    _Float16* __restrict__ out16, float* __restrict__ out32) {
  __shared__ __align__(16) _Float16 As[128 * 40];
  const int tid = threadIdx.x, lane = tid & 31;
  const int w = tid >> 5, wm = w >> 1, wn = w & 1;
  const size_t m0 = (size_t)blockIdx.x * 128;
  const int n0 = blockIdx.y * 128;

  v8f acc[2][4];
#pragma unroll
  for (int mi = 0; mi < 2; ++mi)
#pragma unroll
    for (int ni = 0; ni < 4; ++ni) acc[mi][ni] = (v8f)0.0f;

  // A-chunk mapping for this thread (two 8-half segments per chunk)
  const int r0 = (tid * 2) >> 2, c80 = ((tid * 2) & 3) * 8;
  const int r1 = (tid * 2 + 1) >> 2, c81 = ((tid * 2 + 1) & 3) * 8;

  // B fragment pointer for this lane: advances by 8192 halves per K-chunk,
  // N-subtiles at immediate offsets ni*512 halves.
  const _Float16* bp = Bw + ((size_t)((n0 >> 4) + wn * 4) * 32 + lane) * 16;

  for (int kc = 0; kc < K; kc += 32) {
    *(v8h*)&As[r0 * 40 + c80] = *(const v8h*)(A + (m0 + r0) * lda + kc + c80);
    *(v8h*)&As[r1 * 40 + c81] = *(const v8h*)(A + (m0 + r1) * lda + kc + c81);
    // speculative prefetch of the next A chunk (global_prefetch_b8)
    __builtin_prefetch(A + (m0 + r0) * lda + kc + 32 + c80, 0, 1);
    __syncthreads();
    v16h bf[4];
#pragma unroll
    for (int ni = 0; ni < 4; ++ni) bf[ni] = load_b_frag_sw(bp + ni * 512);
    v16h af0 = load_a_frag(As, wm * 32, lane);
    v16h af1 = load_a_frag(As, wm * 32 + 16, lane);
#pragma unroll
    for (int ni = 0; ni < 4; ++ni) {
      acc[0][ni] = wmma_f16(af0, bf[ni], acc[0][ni]);
      acc[1][ni] = wmma_f16(af1, bf[ni], acc[1][ni]);
    }
    bp += 16 * 512;
    __syncthreads();
  }
  const int rbase = (lane < 16) ? 0 : 8, cloc = lane & 15;
#pragma unroll
  for (int mi = 0; mi < 2; ++mi)
#pragma unroll
    for (int ni = 0; ni < 4; ++ni) {
      int col = n0 + wn * 64 + ni * 16 + cloc;
      float bv = HASBIAS ? bias[col] : 0.0f;
#pragma unroll
      for (int v = 0; v < 8; ++v) {
        size_t row = m0 + wm * 32 + mi * 16 + rbase + v;
        float val = acc[mi][ni][v] + bv;
        if (DOSILU) val = siluf(val);
        if (OUT16) out16[row * HD + col] = (_Float16)val;
        if (OUT32) out32[row * HD + col] = val;
      }
    }
}

// ---------------- fused edge message: A = silu(HiW[ei]+HjW[ej]+EfW[e]+m1_b),
// mij2 = silu(A @ m2 + m2_b), msg[node] = mean over 32 edges.
// Block: 512 thr (16 waves, 4x4), 128 edge rows = 4 complete target nodes.
// m2 weights pre-swizzled; B fragments stream from global (L2-resident).
__global__ __launch_bounds__(512) void edge_msg_kernel(
    const _Float16* __restrict__ HiW, const _Float16* __restrict__ HjW,
    const _Float16* __restrict__ EfW, const float* __restrict__ m1b,
    const _Float16* __restrict__ m2w, const float* __restrict__ m2b,
    _Float16* __restrict__ msg) {
  __shared__ __align__(16) _Float16 As[128 * 40];
  const int tid = threadIdx.x, lane = tid & 31;
  const int w = tid >> 5, wm = w >> 2, wn = w & 3;     // wave tile 32 x 64
  const int e0 = blockIdx.x * 128;

  v8f acc[2][4];
#pragma unroll
  for (int mi = 0; mi < 2; ++mi)
#pragma unroll
    for (int ni = 0; ni < 4; ++ni) acc[mi][ni] = (v8f)0.0f;

  // per-thread A-chunk element group: one v8h segment (128 rows x 32 cols)
  const int r = tid >> 2, c8 = (tid & 3) * 8;
  const int e = e0 + r;
  const int ni_ = e >> 5;                              // target node (ei)
  const int nj_ = ((e >> 10) << 5) + (e & 31);         // source node (ej)

  const _Float16* bp = m2w + ((size_t)(wn * 4) * 32 + lane) * 16;

  for (int kc = 0; kc < HD; kc += 32) {
    { // build A chunk (pre-activation sum + SiLU), packed vector LDS store
      v8h hi = *(const v8h*)(HiW + (size_t)ni_ * HD + kc + c8);
      v8h hj = *(const v8h*)(HjW + (size_t)nj_ * HD + kc + c8);
      v8h ew = *(const v8h*)(EfW + (size_t)e * HD + kc + c8);
      v8h ov;
#pragma unroll
      for (int h = 0; h < 8; ++h) {
        float x = (float)hi[h] + (float)hj[h] + (float)ew[h] + m1b[kc + c8 + h];
        ov[h] = (_Float16)siluf(x);
      }
      *(v8h*)&As[r * 40 + c8] = ov;
    }
    __syncthreads();
    v16h bf[4];
#pragma unroll
    for (int ni = 0; ni < 4; ++ni) bf[ni] = load_b_frag_sw(bp + ni * 512);
    v16h af0 = load_a_frag(As, wm * 32, lane);
    v16h af1 = load_a_frag(As, wm * 32 + 16, lane);
#pragma unroll
    for (int ni = 0; ni < 4; ++ni) {
      acc[0][ni] = wmma_f16(af0, bf[ni], acc[0][ni]);
      acc[1][ni] = wmma_f16(af1, bf[ni], acc[1][ni]);
    }
    bp += 16 * 512;
    __syncthreads();
  }
  // silu + reduce this wave's 32 rows (== one target node)
  const int node = (e0 >> 5) + wm;
  const int cloc = lane & 15;
#pragma unroll
  for (int ni = 0; ni < 4; ++ni) {
    int col = wn * 64 + ni * 16 + cloc;
    float bv = m2b[col], s = 0.0f;
#pragma unroll
    for (int mi = 0; mi < 2; ++mi)
#pragma unroll
      for (int v = 0; v < 8; ++v) s += siluf(acc[mi][ni][v] + bv);
    s += __shfl_down(s, 16, 32);                       // fold rows +8 halves
    if (lane < 16)
      msg[(size_t)node * HD + col] = (_Float16)(s * (1.0f / 32.0f));
  }
}

// ---------------- misc small kernels ----------------
__global__ __launch_bounds__(256) void cat_kernel(const float* __restrict__ nf,
                                                  const _Float16* __restrict__ msg,
                                                  _Float16* __restrict__ cat16) {
  int idx = blockIdx.x * blockDim.x + threadIdx.x;
  if (idx >= NNODE * 2 * HD) return;
  int node = idx >> 9, c = idx & 511;
  cat16[idx] = (c < HD) ? (_Float16)nf[(size_t)node * HD + c]
                        : msg[(size_t)node * HD + (c - HD)];
}

__global__ __launch_bounds__(256) void resid_kernel(float* __restrict__ nf,
                                                    const float* __restrict__ a) {
  int idx = blockIdx.x * blockDim.x + threadIdx.x;
  if (idx < NNODE * HD) nf[idx] += a[idx];
}

__global__ __launch_bounds__(256) void gfeat_kernel(const float* __restrict__ nfln,
                                                    float* __restrict__ gf) {
  int idx = blockIdx.x * blockDim.x + threadIdx.x;   // BG*HD
  if (idx >= BG * HD) return;
  int b = idx >> 8, c = idx & 255;
  float s = 0.0f;
  for (int i = 0; i < NA; ++i) s += nfln[(size_t)(b * NA + i) * HD + c];
  gf[idx] = s * (1.0f / NA);
}

__global__ __launch_bounds__(128) void type_out_kernel(const float* __restrict__ nfln,
                                                       const float* __restrict__ W,
                                                       const float* __restrict__ bias,
                                                       float* __restrict__ out) {
  int node = blockIdx.x, t = threadIdx.x;
  if (t >= TYPED) return;
  const float* x = nfln + (size_t)node * HD;
  float s = bias[t];
  for (int k = 0; k < HD; ++k) s += x[k] * W[k * TYPED + t];
  out[(size_t)node * TYPED + t] = s;
}

__global__ __launch_bounds__(32) void polar_out_kernel(const float* __restrict__ gf,
                                                       const float* __restrict__ W,
                                                       float* __restrict__ out) {
  int idx = blockIdx.x * blockDim.x + threadIdx.x;   // BG*6
  if (idx >= BG * 6) return;
  int b = idx / 6, o = idx - b * 6;
  const float* x = gf + (size_t)b * HD;
  float s = 0.0f;
  for (int k = 0; k < HD; ++k) s += x[k] * W[k * 6 + o];
  out[idx] = s;
}

__global__ __launch_bounds__(256) void frac_out_kernel(const float* __restrict__ nfln,
                                                       const float* __restrict__ W,
                                                       float* __restrict__ out) {
  int idx = blockIdx.x * blockDim.x + threadIdx.x;   // NNODE*3
  if (idx >= NNODE * 3) return;
  int node = idx / 3, o = idx - node * 3;
  const float* x = nfln + (size_t)node * HD;
  float s = 0.0f;
  for (int k = 0; k < HD; ++k) s += x[k] * W[k * 3 + o];
  out[idx] = s;
}

// ---------------- host launcher ----------------
extern "C" void kernel_launch(void* const* d_in, const int* in_sizes, int n_in,
                              void* d_out, int out_size, void* d_ws, size_t ws_size,
                              hipStream_t stream) {
  const float* t     = (const float*)d_in[0];
  const float* atomt = (const float*)d_in[1];
  const float* frac  = (const float*)d_in[2];
  const float* lpol  = (const float*)d_in[3];
  // d_in[4] node2graph: structure is deterministic (repeat), not needed
  const float* Ws = (const float*)d_in[5];  const float* bs = (const float*)d_in[6];
  const float* Wn = (const float*)d_in[7];  const float* bn = (const float*)d_in[8];
  const float* lng = (const float*)d_in[9]; const float* lnb = (const float*)d_in[10];
  const float* m1W = (const float*)d_in[11];const float* m1b = (const float*)d_in[12];
  const float* m2W = (const float*)d_in[13];const float* m2b = (const float*)d_in[14];
  const float* a1W = (const float*)d_in[15];const float* a1b = (const float*)d_in[16];
  const float* a2W = (const float*)d_in[17];const float* a2b = (const float*)d_in[18];
  const float* flng = (const float*)d_in[19];const float* flnb = (const float*)d_in[20];
  const float* typW = (const float*)d_in[21];const float* typb = (const float*)d_in[22];
  const float* polW = (const float*)d_in[23];const float* frcW = (const float*)d_in[24];
  float* out = (float*)d_out;

  // workspace carve-up (256B aligned)
  char* ws = (char*)d_ws;
  size_t off = 0;
  auto carve = [&](size_t bytes) { size_t o = off; off = (off + bytes + 255) & ~(size_t)255; return o; };
  float*    nf      = (float*)   (ws + carve((size_t)NNODE * HD * 4));
  float*    scrF    = (float*)   (ws + carve((size_t)NNODE * HD * 4)); // te / aout / nf_ln
  float*    temb    = (float*)   (ws + carve((size_t)BG * TDIM * 4));
  float*    gf      = (float*)   (ws + carve((size_t)BG * HD * 4));
  _Float16* h16     = (_Float16*)(ws + carve((size_t)NNODE * HD * 2));
  _Float16* HiW16   = (_Float16*)(ws + carve((size_t)NNODE * HD * 2));
  _Float16* HjW16   = (_Float16*)(ws + carve((size_t)NNODE * HD * 2));
  _Float16* ef16    = (_Float16*)(ws + carve((size_t)NEDGE * EDGEP * 2));
  _Float16* EfW16   = (_Float16*)(ws + carve((size_t)NEDGE * HD * 2));
  _Float16* msg16   = (_Float16*)(ws + carve((size_t)NNODE * HD * 2));
  _Float16* cat16   = (_Float16*)(ws + carve((size_t)NNODE * 2 * HD * 2));
  _Float16* tmp16   = (_Float16*)(ws + carve((size_t)NNODE * HD * 2));
  const size_t LHALVES = (size_t)(256 + 256 + 96 + 256 + 512 + 256) * HD; // per layer
  _Float16* w16     = (_Float16*)(ws + carve(NLAY * LHALVES * 2));
  const size_t oWa = 0, oWb = 256 * HD, oWc = 512 * HD, oM2 = oWc + 96 * HD,
               oA1 = oM2 + 256 * HD, oA2 = oA1 + 512 * HD;

  auto grid1 = [](size_t n, int bs_) { return dim3((unsigned)((n + bs_ - 1) / bs_)); };

  // 1) cast weights to f16, pre-swizzled into WMMA B-fragment order
  //    (Wc rows padded 66->96 with zeros; ef cols also padded)
  for (int l = 0; l < NLAY; ++l) {
    _Float16* wl = w16 + (size_t)l * LHALVES;
    castswz_kernel<<<grid1(256*HD,256),256,0,stream>>>(m1W + (size_t)l*578*HD,           wl+oWa, 256,256,HD);
    castswz_kernel<<<grid1(256*HD,256),256,0,stream>>>(m1W + (size_t)l*578*HD + 256*HD,  wl+oWb, 256,256,HD);
    castswz_kernel<<<grid1(96*HD,256), 256,0,stream>>>(m1W + (size_t)l*578*HD + 512*HD,  wl+oWc,  66, 96,HD);
    castswz_kernel<<<grid1(256*HD,256),256,0,stream>>>(m2W + (size_t)l*HD*HD,            wl+oM2, 256,256,HD);
    castswz_kernel<<<grid1(512*HD,256),256,0,stream>>>(a1W + (size_t)l*2*HD*HD,          wl+oA1, 512,512,HD);
    castswz_kernel<<<grid1(256*HD,256),256,0,stream>>>(a2W + (size_t)l*HD*HD,            wl+oA2, 256,256,HD);
  }

  // 2) embeddings
  time_emb_kernel<<<BG, TDIM, 0, stream>>>(t, temb);
  type_emb_kernel<<<NNODE, HD, 0, stream>>>(atomt, Ws, bs, scrF);
  node_emb_kernel<<<NNODE, HD, 0, stream>>>(scrF, temb, Wn, bn, nf);

  // 3) edge features
  edge_feat_kernel<<<grid1(NEDGE,256), 256, 0, stream>>>(frac, lpol, ef16);

  // 4) GNN layers
  for (int l = 0; l < NLAY; ++l) {
    _Float16* wl = w16 + (size_t)l * LHALVES;
    ln_kernel<<<NNODE, HD, 0, stream>>>(nf, lng + l*HD, lnb + l*HD, h16, nullptr);
    // node pre-projections of m1 (concat split): h@Wa, h@Wb
    gemm_wmma_kernel<false,false,true,false><<<dim3(NNODE/128,2),256,0,stream>>>(
        h16, HD, HD, wl+oWa, nullptr, HiW16, nullptr);
    gemm_wmma_kernel<false,false,true,false><<<dim3(NNODE/128,2),256,0,stream>>>(
        h16, HD, HD, wl+oWb, nullptr, HjW16, nullptr);
    // ef @ Wc  (K padded to 96)
    gemm_wmma_kernel<false,false,true,false><<<dim3(NEDGE/128,2),256,0,stream>>>(
        ef16, EDGEP, EDGEP, wl+oWc, nullptr, EfW16, nullptr);
    // fused: silu(Hi+Hj+EfW+b) @ m2 -> silu -> mean over 32 edges per node
    edge_msg_kernel<<<NEDGE/128, 512, 0, stream>>>(HiW16, HjW16, EfW16, m1b + l*HD,
                                                   wl+oM2, m2b + l*HD, msg16);
    // node update MLP with residual
    cat_kernel<<<grid1((size_t)NNODE*2*HD,256),256,0,stream>>>(nf, msg16, cat16);
    gemm_wmma_kernel<true,true,true,false><<<dim3(NNODE/128,2),256,0,stream>>>(
        cat16, 2*HD, 2*HD, wl+oA1, a1b + l*HD, tmp16, nullptr);
    gemm_wmma_kernel<true,true,false,true><<<dim3(NNODE/128,2),256,0,stream>>>(
        tmp16, HD, HD, wl+oA2, a2b + l*HD, nullptr, scrF);
    resid_kernel<<<grid1((size_t)NNODE*HD,256),256,0,stream>>>(nf, scrF);
  }

  // 5) final LN + readouts
  ln_kernel<<<NNODE, HD, 0, stream>>>(nf, flng, flnb, nullptr, scrF);
  type_out_kernel<<<NNODE, 128, 0, stream>>>(scrF, typW, typb, out);                 // [4096,100]
  gfeat_kernel<<<grid1((size_t)BG*HD,256),256,0,stream>>>(scrF, gf);
  polar_out_kernel<<<grid1((size_t)BG*6,32),32,0,stream>>>(gf, polW, out + (size_t)NNODE*TYPED);
  frac_out_kernel<<<grid1((size_t)NNODE*3,256),256,0,stream>>>(scrF, frcW,
                                              out + (size_t)NNODE*TYPED + (size_t)BG*6);
}
